// BinaryMLP_48404281426021
// MI455X (gfx1250) — compile-verified, matrix operand order checked by reading, not run on hardware
//
#include <hip/hip_runtime.h>

// ---------------------------------------------------------------------------
// Fused binary-MLP forward for MI455X (gfx1250, wave32, WMMA).
//   h = relu(x @ sign(W1)^T + b1); h = relu(h @ sign(W2)^T + b2);
//   h = relu(h @ sign(W3)^T + b3); out = h @ W4^T + b4
// Memory-bound on streaming x (411 MB @ 23.3 TB/s ~= 19 us floor), so all four
// layers are fused: one workgroup stages 128 rows of x + all (binarized f16)
// weights into LDS and chains layers per-wave through LDS ping-pong buffers.
// Matrix math uses V_WMMA_F32_16X16X32_F16 (f32 accumulate).
// ---------------------------------------------------------------------------

typedef _Float16 v8h  __attribute__((ext_vector_type(8)));
typedef _Float16 v16h __attribute__((ext_vector_type(16)));
typedef float    v8f  __attribute__((ext_vector_type(8)));

#define THREADS   256
#define WG_ROWS   128          // batch rows per workgroup (8 waves x 16-row tiles)
#define D_IN      196
#define K1_STEPS  7            // ceil(196/32); padded K = 224
#define XSTR      232          // LDS row stride (halfs) for x tile & W1 (bank rotate)
#define HDIM      128
#define K2_STEPS  4            // 128/32
#define HSTR      136          // LDS row stride (halfs) for h / W2 / W3 / W4
#define D_OUT     10

#define XT_HALFS  (WG_ROWS * XSTR)   // 29696
#define W1_HALFS  (HDIM * XSTR)      // 29696
#define W2_HALFS  (HDIM * HSTR)      // 17408 (also W3)
#define W4_HALFS  (16 * HSTR)        // 2176 (rows 10..15 zero-padded)
#define H_HALFS   (WG_ROWS * HSTR)   // 17408 (two ping-pong buffers)

#define SMEM_HALFS (XT_HALFS + W1_HALFS + 2 * W2_HALFS + W4_HALFS + 2 * H_HALFS)
#define SMEM_BYTES ((size_t)SMEM_HALFS * 2 + (size_t)(3 * 128 + 16) * 4)

// Load one 16x32 f16 WMMA operand fragment from LDS.
// Per ISA 7.12.2: lane L<16 -> row (lane&15), K chunks [k0..k0+7] and
// [k0+16..k0+23]; lanes 16..31 take the +8 half-offset chunks. Each chunk is
// 8 contiguous halfs = one ds_load_b128.
__device__ __forceinline__ v16h ldsFrag(const _Float16* base, int row, int stride,
                                        int khalf, int lane) {
  const int off = row * stride + khalf + ((lane & 16) ? 8 : 0);
  v8h lo = *(const v8h*)(base + off);
  v8h hi = *(const v8h*)(base + off + 16);
  v16h r;
#pragma unroll
  for (int i = 0; i < 8; ++i) { r[i] = lo[i]; r[i + 8] = hi[i]; }
  return r;
}

__device__ __forceinline__ v8f wmma_f16(v16h a, v16h b, v8f c) {
  // 8 args: (neg_a, A, neg_b, B, c_mod, C, reuse_a, reuse_b)
  return __builtin_amdgcn_wmma_f32_16x16x32_f16(false, a, false, b, (short)0, c,
                                                false, false);
}

// One hidden layer for this wave's 16-row tile: act_out = relu(A @ W^T + bias).
// Output is 128 wide -> 8 N-tiles of 16. A fragments hoisted across N-tiles.
template <int KSTEPS>
__device__ __forceinline__ void hidden_layer(const _Float16* act_in, int in_stride,
                                             const _Float16* wmat, int w_stride,
                                             const float* bias,
                                             _Float16* act_out, int out_stride,
                                             int wave, int lane) {
  const int mrow = wave * 16 + (lane & 15);
  v16h afr[KSTEPS];
#pragma unroll
  for (int ks = 0; ks < KSTEPS; ++ks)
    afr[ks] = ldsFrag(act_in, mrow, in_stride, ks * 32, lane);

#pragma unroll
  for (int nt = 0; nt < 8; ++nt) {
    v8f acc = {};
#pragma unroll
    for (int ks = 0; ks < KSTEPS; ++ks) {
      v16h bfr = ldsFrag(wmat, nt * 16 + (lane & 15), w_stride, ks * 32, lane);
      acc = wmma_f16(afr[ks], bfr, acc);
    }
    // Epilogue: D layout VGPR r -> rows {r, r+8}, N = lane&15 per half-wave.
    const int col = nt * 16 + (lane & 15);
    const float bv = bias[col];
    const int rb = wave * 16 + ((lane & 16) ? 8 : 0);
#pragma unroll
    for (int r = 0; r < 8; ++r) {
      float v = acc[r] + bv;
      v = v > 0.f ? v : 0.f;  // ReLU
      act_out[(rb + r) * out_stride + col] = (_Float16)v;
    }
  }
}

__global__ void __launch_bounds__(THREADS)
binary_mlp_fused(const float* __restrict__ x,
                 const float* __restrict__ W1g, const float* __restrict__ b1g,
                 const float* __restrict__ W2g, const float* __restrict__ b2g,
                 const float* __restrict__ W3g, const float* __restrict__ b3g,
                 const float* __restrict__ W4g, const float* __restrict__ b4g,
                 float* __restrict__ out, int nrows) {
  extern __shared__ __align__(16) unsigned char smem_raw[];
  _Float16* xt = (_Float16*)smem_raw;
  _Float16* w1 = xt + XT_HALFS;
  _Float16* w2 = w1 + W1_HALFS;
  _Float16* w3 = w2 + W2_HALFS;
  _Float16* w4 = w3 + W2_HALFS;
  _Float16* h1 = w4 + W4_HALFS;
  _Float16* h2 = h1 + H_HALFS;
  float* bias1 = (float*)(h2 + H_HALFS);
  float* bias2 = bias1 + 128;
  float* bias3 = bias2 + 128;
  float* bias4 = bias3 + 128;

  const int tid = threadIdx.x;
  const int lane = tid & 31;
  const int wave = tid >> 5;
  const int row0 = blockIdx.x * WG_ROWS;

  // ---- stage binarized W1 (f16, zero K-pad 196..231) ----
  for (int i = tid; i < W1_HALFS; i += THREADS) {
    const int n = i / XSTR, k = i - n * XSTR;
    float v = 0.f;
    if (k < D_IN) {
      const float w = W1g[n * D_IN + k];
      v = (w > 0.f) ? 1.f : ((w < 0.f) ? -1.f : 0.f);
    }
    w1[i] = (_Float16)v;
  }
  // ---- stage binarized W2 / W3 ----
  for (int i = tid; i < W2_HALFS; i += THREADS) {
    const int n = i / HSTR, k = i - n * HSTR;
    float v2 = 0.f, v3 = 0.f;
    if (k < HDIM) {
      const float a = W2g[n * HDIM + k];
      v2 = (a > 0.f) ? 1.f : ((a < 0.f) ? -1.f : 0.f);
      const float b = W3g[n * HDIM + k];
      v3 = (b > 0.f) ? 1.f : ((b < 0.f) ? -1.f : 0.f);
    }
    w2[i] = (_Float16)v2;
    w3[i] = (_Float16)v3;
  }
  // ---- stage full-precision head W4 (rows padded 10->16) ----
  for (int i = tid; i < W4_HALFS; i += THREADS) {
    const int n = i / HSTR, k = i - n * HSTR;
    const float v = (n < D_OUT && k < HDIM) ? W4g[n * HDIM + k] : 0.f;
    w4[i] = (_Float16)v;
  }
  if (tid < 128) {
    bias1[tid] = b1g[tid];
    bias2[tid] = b2g[tid];
    bias3[tid] = b3g[tid];
  }
  if (tid < 16) bias4[tid] = (tid < D_OUT) ? b4g[tid] : 0.f;

  // ---- stage x tile f32 -> f16 (coalesced, zero K-pad) ----
  for (int i = tid; i < XT_HALFS; i += THREADS) {
    const int r = i / XSTR, k = i - r * XSTR;
    const int gr = row0 + r;
    const float v = (k < D_IN && gr < nrows) ? x[(size_t)gr * D_IN + k] : 0.f;
    xt[i] = (_Float16)v;
  }
  __syncthreads();

  // Layers chain per-wave (each wave reads only the h rows it wrote).
  hidden_layer<K1_STEPS>(xt, XSTR, w1, XSTR, bias1, h1, HSTR, wave, lane);
  hidden_layer<K2_STEPS>(h1, HSTR, w2, HSTR, bias2, h2, HSTR, wave, lane);
  hidden_layer<K2_STEPS>(h2, HSTR, w3, HSTR, bias3, h1, HSTR, wave, lane);

  // ---- head: out[16 rows x 10] = h1 @ W4^T + b4 (single N-tile) ----
  {
    const int mrow = wave * 16 + (lane & 15);
    v16h afr[K2_STEPS];
#pragma unroll
    for (int ks = 0; ks < K2_STEPS; ++ks)
      afr[ks] = ldsFrag(h1, mrow, HSTR, ks * 32, lane);
    v8f acc = {};
#pragma unroll
    for (int ks = 0; ks < K2_STEPS; ++ks)
      acc = wmma_f16(afr[ks], ldsFrag(w4, lane & 15, HSTR, ks * 32, lane), acc);

    const int col = lane & 15;
    const int rb = row0 + wave * 16 + ((lane & 16) ? 8 : 0);
    if (col < D_OUT) {
      const float bv = bias4[col];
#pragma unroll
      for (int r = 0; r < 8; ++r) {
        const int gr = rb + r;
        if (gr < nrows) out[(size_t)gr * D_OUT + col] = acc[r] + bv;
      }
    }
  }
}

extern "C" void kernel_launch(void* const* d_in, const int* in_sizes, int n_in,
                              void* d_out, int out_size, void* d_ws, size_t ws_size,
                              hipStream_t stream) {
  (void)n_in; (void)out_size; (void)d_ws; (void)ws_size;
  const float* x  = (const float*)d_in[0];
  const float* W1 = (const float*)d_in[1];
  const float* b1 = (const float*)d_in[2];
  const float* W2 = (const float*)d_in[3];
  const float* b2 = (const float*)d_in[4];
  const float* W3 = (const float*)d_in[5];
  const float* b3 = (const float*)d_in[6];
  const float* W4 = (const float*)d_in[7];
  const float* b4 = (const float*)d_in[8];
  float* out = (float*)d_out;

  const int nrows = in_sizes[0] / D_IN;          // 524288
  const int grid = (nrows + WG_ROWS - 1) / WG_ROWS;
  if (grid <= 0) return;

  // Allow > default dynamic LDS (gfx1250 WGP has 320 KB; we use ~258 KB).
  (void)hipFuncSetAttribute((const void*)binary_mlp_fused,
                            hipFuncAttributeMaxDynamicSharedMemorySize,
                            (int)SMEM_BYTES);

  binary_mlp_fused<<<grid, THREADS, SMEM_BYTES, stream>>>(
      x, W1, b1, W2, b2, W3, b3, W4, b4, out, nrows);
}
